// TripletLoss_50749333569671
// MI455X (gfx1250) — matrix operand order repaired and checked by reading, not compiled, not run on hardware
//
#include <hip/hip_runtime.h>

typedef __attribute__((ext_vector_type(16))) __bf16 v16bf;
typedef __attribute__((ext_vector_type(8)))  __bf16 v8bf;
typedef __attribute__((ext_vector_type(8)))  float  v8f;

#define DIM 128            // feature dimension (fixed by reference)
#define MARGIN 0.3f
#define EPS 1e-6f
#define KSPLIT 4           // split-K for GEMM1 (occupancy for the 256MB mask stream)

static __device__ __forceinline__ unsigned short f32_to_bf16(float f) {
    unsigned u = __float_as_uint(f);
    unsigned rnd = 0x7FFFu + ((u >> 16) & 1u);   // round-to-nearest-even
    return (unsigned short)((u + rnd) >> 16);
}

// --- CDNA5 async Global->LDS DMA (ASYNCcnt path) ---------------------------
// Copies 64 contiguous bytes. inst_offset is added to BOTH the LDS and the
// global address, so one base pair covers 4 x B128.
static __device__ __forceinline__ void stage_b64_async(const unsigned short* g,
                                                       unsigned short* l) {
    unsigned long long ga = (unsigned long long)(uintptr_t)g;
    __attribute__((address_space(3))) unsigned short* la =
        (__attribute__((address_space(3))) unsigned short*)l;
    asm volatile("global_load_async_to_lds_b128 %0, %1, off\n\t"
                 "global_load_async_to_lds_b128 %0, %1, off offset:16\n\t"
                 "global_load_async_to_lds_b128 %0, %1, off offset:32\n\t"
                 "global_load_async_to_lds_b128 %0, %1, off offset:48"
                 :: "v"(la), "v"(ga) : "memory");
}
static __device__ __forceinline__ void wait_async0() {
    asm volatile("s_wait_asynccnt 0x0" ::: "memory");
}

// ---------------------------------------------------------------------------
// Kernel 0: re-init accumulators (deterministic every call).
// ---------------------------------------------------------------------------
__global__ void tl_init(float* __restrict__ csum, int* __restrict__ counts,
                        unsigned long long* __restrict__ argm,
                        float* __restrict__ out, int n) {
    int i = blockIdx.x * blockDim.x + threadIdx.x;
    if (i < n * DIM) csum[i] = 0.0f;
    if (i < n) { counts[i] = 0; argm[i] = 0xFFFFFFFFFFFFFFFFull; }
    if (i == 0) out[0] = 0.0f;
}

// ---------------------------------------------------------------------------
// Kernel 1: inputs(f32,[n][128]) -> bf16 row-major + bf16 col-major, and
// x2[j] = ||inputs[j]||^2. One 128-thread block per row j.
// ---------------------------------------------------------------------------
__global__ __launch_bounds__(128) void tl_prep(
    const float* __restrict__ x,
    unsigned short* __restrict__ x_bf,     // [n][128]
    unsigned short* __restrict__ xT_bf,    // [128][n]
    float* __restrict__ x2, int n)
{
    const int j = blockIdx.x;
    const int t = threadIdx.x;
    float v = x[(size_t)j * DIM + t];
    unsigned short b = f32_to_bf16(v);
    x_bf[(size_t)j * DIM + t] = b;
    xT_bf[(size_t)t * n + j]  = b;

    float s = v * v;
    s += __shfl_xor(s, 1);  s += __shfl_xor(s, 2);
    s += __shfl_xor(s, 4);  s += __shfl_xor(s, 8);
    s += __shfl_xor(s, 16);
    __shared__ float parts[4];
    if ((threadIdx.x & 31) == 0) parts[threadIdx.x >> 5] = s;
    __syncthreads();
    if (threadIdx.x == 0) x2[j] = parts[0] + parts[1] + parts[2] + parts[3];
}

// ---------------------------------------------------------------------------
// Kernel 2: GEMM1 partial sums:  csum += maskf(K-slice) @ inputs,
// counts += row-nonzeros(K-slice).  M=n, N=128, K-slice = n/KSPLIT.
// 4 waves/block; wave: 16 rows x 128 cols. B tile (32x128 bf16, 8KB) is
// double-buffered in LDS via async Global->LDS DMA, overlapped with WMMA.
// Mask int32 -> bf16 {0,1} fused into the A-fragment load (exact in bf16).
// ---------------------------------------------------------------------------
__global__ __launch_bounds__(128) void tl_gemm1(
    const int* __restrict__ sim,              // [n][n]
    const unsigned short* __restrict__ xT,    // [128][n] bf16
    float* __restrict__ csum,                 // [n][128] (atomic accumulate)
    int* __restrict__ counts,                 // [n]      (atomic accumulate)
    int n, int kspan)
{
    const int lane = threadIdx.x & 31;
    const int wave = threadIdx.x >> 5;
    const int m0   = blockIdx.x * 64 + wave * 16;
    const int l15  = lane & 15;
    const int hl   = lane >> 4;        // 0 or 1 (half of wave)
    const int klo  = hl ? 16 : 0;      // A-fragment K half (contiguous, matches B)
    const int kb   = hl ? 16 : 0;      // B-fragment K half

    __shared__ alignas(64) unsigned short ldsB[2][DIM * 32];  // 2 x 8KB

    const int kbeg = blockIdx.y * kspan;
    const int kend = kbeg + kspan;
    const int t128 = threadIdx.x;      // 128 threads stage 128 columns x 64B

    v8f c[8];
    const v8f vzero = {0.f,0.f,0.f,0.f,0.f,0.f,0.f,0.f};
#pragma unroll
    for (int t = 0; t < 8; ++t) c[t] = vzero;

    int cnt = 0;
    const int* mrow = sim + (size_t)(m0 + l15) * (size_t)n;

    // prologue: stage first K-slice of B
    stage_b64_async(xT + (size_t)t128 * n + kbeg, &ldsB[0][t128 * 32]);

    int cur = 0;
    for (int k0 = kbeg; k0 < kend; k0 += 32) {
        wait_async0();          // own async stores into ldsB[cur] complete
        __syncthreads();        // all waves staged; prior reads of other buf done
        if (k0 + 32 < kend)     // overlap next slice's DMA with this slice's WMMAs
            stage_b64_async(xT + (size_t)t128 * n + (k0 + 32),
                            &ldsB[cur ^ 1][t128 * 32]);

        // ---- A fragment: 16 mask ints -> bf16 0/1, count nonzeros ----
        const int* mp = mrow + k0 + klo;
        int av[16];
        *(int4*)(&av[0])  = *(const int4*)(mp + 0);
        *(int4*)(&av[4])  = *(const int4*)(mp + 4);
        *(int4*)(&av[8])  = *(const int4*)(mp + 8);
        *(int4*)(&av[12]) = *(const int4*)(mp + 12);
        union { v16bf v; unsigned short u[16]; } A;
#pragma unroll
        for (int i = 0; i < 16; ++i) {
            int nz = (av[i] != 0);
            cnt += nz;
            A.u[i] = nz ? (unsigned short)0x3F80 : (unsigned short)0;
        }
        if (k0 + 32 < kend) __builtin_prefetch(mp + 32, 0, 1);  // global_prefetch_b8

        // ---- load all 8 B fragments from LDS, then 8 WMMAs ----
        v16bf B[8];
#pragma unroll
        for (int t = 0; t < 8; ++t)
            B[t] = *reinterpret_cast<const v16bf*>(
                &ldsB[cur][(16 * t + l15) * 32 + kb]);
#pragma unroll
        for (int t = 0; t < 8; ++t)
            c[t] = __builtin_amdgcn_wmma_f32_16x16x32_bf16(
                false, A.v, false, B[t], (short)0, c[t], false, false);

        cur ^= 1;
    }

    // lanes L and L+16 cover complementary K of row m0+l15 -> slice count
    cnt += __shfl_xor(cnt, 16);
    if (hl == 0) atomicAdd(&counts[m0 + l15], cnt);

#pragma unroll
    for (int r = 0; r < 8; ++r) {
        const int row = m0 + r + 8 * hl;
#pragma unroll
        for (int t = 0; t < 8; ++t)
            atomicAdd(&csum[(size_t)row * DIM + 16 * t + l15], c[t][r]);
    }
}

// ---------------------------------------------------------------------------
// Kernel 3: centers = csum / counts; also bf16 copy and c2[i]=||centers_i||^2.
// ---------------------------------------------------------------------------
__global__ __launch_bounds__(128) void tl_post(
    const float* __restrict__ csum, const int* __restrict__ counts,
    float* __restrict__ centers, unsigned short* __restrict__ centers_bf,
    float* __restrict__ c2, int n)
{
    const int j = blockIdx.x;
    const int t = threadIdx.x;
    float v = csum[(size_t)j * DIM + t] / (float)counts[j];
    centers[(size_t)j * DIM + t]    = v;
    centers_bf[(size_t)j * DIM + t] = f32_to_bf16(v);

    float s = v * v;
    s += __shfl_xor(s, 1);  s += __shfl_xor(s, 2);
    s += __shfl_xor(s, 4);  s += __shfl_xor(s, 8);
    s += __shfl_xor(s, 16);
    __shared__ float parts[4];
    if ((threadIdx.x & 31) == 0) parts[threadIdx.x >> 5] = s;
    __syncthreads();
    if (threadIdx.x == 0) c2[j] = parts[0] + parts[1] + parts[2] + parts[3];
}

// ---------------------------------------------------------------------------
// Kernel 4: GEMM2 + fused masked argmin.
// d2[i,j] = c2[i] + x2[j] - 2*(centers_i . input_j); +inf where sim!=0;
// per-row min as packed (f32bits<<32 | col) keys -> shuffle reduce ->
// one atomicMin(u64) per (row, 128-col block). B tile (128x128 bf16, 32KB)
// staged once per block via async Global->LDS DMA, shared by all 8 waves.
// ---------------------------------------------------------------------------
__global__ __launch_bounds__(256) void tl_gemm2_argmin(
    const unsigned short* __restrict__ centers_bf, // [n][128]
    const unsigned short* __restrict__ x_bf,       // [n][128]
    const float* __restrict__ c2,
    const float* __restrict__ x2,
    const int* __restrict__ sim,
    unsigned long long* __restrict__ argm, int n)
{
    const int lane = threadIdx.x & 31;
    const int wave = threadIdx.x >> 5;
    const int l15  = lane & 15;
    const int hl   = lane >> 4;
    const int m0   = blockIdx.y * 128 + wave * 16;
    const int j0   = blockIdx.x * 128;
    const int klo  = hl ? 16 : 0;
    const int kb   = hl ? 16 : 0;

    __shared__ alignas(64) unsigned short ldsB[DIM * DIM];   // 32KB B tile

    // stage B: 256 threads, each 128B (two 64B async chunks)
    {
        const int col  = threadIdx.x >> 1;
        const int half = (threadIdx.x & 1) * 64;
        const unsigned short* src = x_bf + (size_t)(j0 + col) * DIM + half;
        unsigned short* dst = &ldsB[col * DIM + half];
        stage_b64_async(src, dst);
        stage_b64_async(src + 32, dst + 32);
    }
    wait_async0();
    __syncthreads();

    v8f c[8];
    const v8f vzero = {0.f,0.f,0.f,0.f,0.f,0.f,0.f,0.f};
#pragma unroll
    for (int t = 0; t < 8; ++t) c[t] = vzero;

#pragma unroll
    for (int k0 = 0; k0 < DIM; k0 += 32) {
        const unsigned short* ap =
            centers_bf + (size_t)(m0 + l15) * DIM + (k0 + klo);
        union { v16bf v; v8bf h[2]; } A;
        A.h[0] = *reinterpret_cast<const v8bf*>(ap);
        A.h[1] = *reinterpret_cast<const v8bf*>(ap + 8);
        v16bf B[8];
#pragma unroll
        for (int t = 0; t < 8; ++t)
            B[t] = *reinterpret_cast<const v16bf*>(
                &ldsB[(16 * t + l15) * DIM + k0 + kb]);
#pragma unroll
        for (int t = 0; t < 8; ++t)
            c[t] = __builtin_amdgcn_wmma_f32_16x16x32_bf16(
                false, A.v, false, B[t], (short)0, c[t], false, false);
    }

    float c2r[8];
#pragma unroll
    for (int r = 0; r < 8; ++r) c2r[r] = c2[m0 + r + 8 * hl];

    unsigned long long kmin[8];
#pragma unroll
    for (int r = 0; r < 8; ++r) kmin[r] = 0xFFFFFFFFFFFFFFFFull;

#pragma unroll
    for (int t = 0; t < 8; ++t) {
        const int col = j0 + 16 * t + l15;
        const float x2c = x2[col];
#pragma unroll
        for (int r = 0; r < 8; ++r) {
            const int row = m0 + r + 8 * hl;
            float d2 = c2r[r] + x2c - 2.0f * c[t][r];
            d2 = fmaxf(d2, 0.0f);
            unsigned bits = __float_as_uint(d2);          // monotone for d2>=0
            if (sim[(size_t)row * (size_t)n + col] != 0)
                bits = 0x7F800000u;                        // +inf: exclude positives
            unsigned long long key =
                ((unsigned long long)bits << 32) | (unsigned)col;
            kmin[r] = key < kmin[r] ? key : kmin[r];
        }
    }
#pragma unroll
    for (int r = 0; r < 8; ++r) {
        unsigned long long k = kmin[r];
        unsigned long long o;
        o = __shfl_xor(k, 1); k = o < k ? o : k;
        o = __shfl_xor(k, 2); k = o < k ? o : k;
        o = __shfl_xor(k, 4); k = o < k ? o : k;
        o = __shfl_xor(k, 8); k = o < k ? o : k;
        if (l15 == r) atomicMin(&argm[m0 + r + 8 * hl], k);
    }
}

// ---------------------------------------------------------------------------
// Kernel 5: finalize. One wave per anchor row.
// ---------------------------------------------------------------------------
__global__ __launch_bounds__(256) void tl_finalize(
    const float* __restrict__ centers, const float* __restrict__ x,
    const unsigned long long* __restrict__ argm,
    float* __restrict__ out, int n)
{
    const int lane = threadIdx.x & 31;
    const int wave = threadIdx.x >> 5;
    const int i = blockIdx.x * 8 + wave;
    const unsigned j = (unsigned)(argm[i] & 0xFFFFFFFFull);

    const float4 cv = *(const float4*)(centers + (size_t)i * DIM + lane * 4);
    const float4 av = *(const float4*)(x       + (size_t)i * DIM + lane * 4);
    const float4 gv = *(const float4*)(x       + (size_t)j * DIM + lane * 4);

    float dap = 0.f, dan = 0.f;
    {
        float e;
        e = cv.x - av.x + EPS; dap += e * e;
        e = cv.y - av.y + EPS; dap += e * e;
        e = cv.z - av.z + EPS; dap += e * e;
        e = cv.w - av.w + EPS; dap += e * e;
        e = cv.x - gv.x + EPS; dan += e * e;
        e = cv.y - gv.y + EPS; dan += e * e;
        e = cv.z - gv.z + EPS; dan += e * e;
        e = cv.w - gv.w + EPS; dan += e * e;
    }
    dap += __shfl_xor(dap, 1);  dan += __shfl_xor(dan, 1);
    dap += __shfl_xor(dap, 2);  dan += __shfl_xor(dan, 2);
    dap += __shfl_xor(dap, 4);  dan += __shfl_xor(dan, 4);
    dap += __shfl_xor(dap, 8);  dan += __shfl_xor(dan, 8);
    dap += __shfl_xor(dap, 16); dan += __shfl_xor(dan, 16);

    if (lane == 0) {
        float l = fmaxf(sqrtf(dap) - sqrtf(dan) + MARGIN, 0.0f);
        atomicAdd(out, l * (1.0f / (float)n));
    }
}

// ---------------------------------------------------------------------------
extern "C" void kernel_launch(void* const* d_in, const int* in_sizes, int n_in,
                              void* d_out, int out_size, void* d_ws, size_t ws_size,
                              hipStream_t stream) {
    const float* x   = (const float*)d_in[0];   // (n, 128) f32
    const int*   sim = (const int*)d_in[1];     // (n, n) int32
    float* out = (float*)d_out;
    const int n = in_sizes[0] / DIM;            // 8192

    // workspace layout (bytes)
    char* ws = (char*)d_ws;
    unsigned short* x_bf       = (unsigned short*)(ws);                      // n*128 bf16
    unsigned short* xT_bf      = (unsigned short*)(ws + (size_t)n * 256);    // 128*n bf16
    float*          centers    = (float*)        (ws + (size_t)n * 512);     // n*128 f32
    unsigned short* centers_bf = (unsigned short*)(ws + (size_t)n * 1024);   // n*128 bf16
    float*          csum       = (float*)        (ws + (size_t)n * 1280);    // n*128 f32
    float*          x2         = (float*)        (ws + (size_t)n * 1792);    // n f32
    float*          c2         = (float*)        (ws + (size_t)n * 1792 + (size_t)n * 4);
    int*            counts     = (int*)          (ws + (size_t)n * 1792 + (size_t)n * 8);
    unsigned long long* argm   = (unsigned long long*)(ws + (size_t)n * 1792 + (size_t)n * 12);

    tl_init<<<(n * DIM + 255) / 256, 256, 0, stream>>>(csum, counts, argm, out, n);
    tl_prep<<<n, 128, 0, stream>>>(x, x_bf, xT_bf, x2, n);
    dim3 g1(n / 64, KSPLIT);
    tl_gemm1<<<g1, 128, 0, stream>>>(sim, xT_bf, csum, counts, n, n / KSPLIT);
    tl_post<<<n, 128, 0, stream>>>(csum, counts, centers, centers_bf, c2, n);
    dim3 g2(n / 128, n / 128);
    tl_gemm2_argmin<<<g2, 256, 0, stream>>>(centers_bf, x_bf, c2, x2, sim, argm, n);
    tl_finalize<<<n / 8, 256, 0, stream>>>(centers, x, argm, out, n);
}
